// Head_11914239279738
// MI455X (gfx1250) — matrix operand order repaired and checked by reading, not compiled, not run on hardware
//
#include <hip/hip_runtime.h>

#define BATCH 8
#define SEQ   2048
#define CEMB  1024
#define DHEAD 64

#define KC        64                  // K-chunk staged in LDS per pipeline stage
#define NCHUNK    (CEMB / KC)         // 16
#define WROW      72                  // LDS row stride in bf16 elems (144B = 128B row + 16B pad)

typedef __attribute__((ext_vector_type(16))) __bf16 v16bf;
typedef __attribute__((ext_vector_type(8)))  float  v8f;
typedef __attribute__((ext_vector_type(2)))  float  f32x2;
typedef __attribute__((ext_vector_type(2)))  __bf16 bf16x2;
typedef __attribute__((ext_vector_type(4)))  unsigned v4u;
typedef __attribute__((ext_vector_type(8)))  int      v8i;
typedef __attribute__((ext_vector_type(4)))  int      v4i;

#if defined(__has_builtin)
#  if __has_builtin(__builtin_amdgcn_tensor_load_to_lds) && \
      __has_builtin(__builtin_amdgcn_s_wait_tensorcnt)
#    define USE_TDM 1
#  endif
#endif
#ifndef USE_TDM
#  define USE_TDM 0
#endif

union Frag { unsigned u[8]; uint4 q[2]; v16bf v; };

__device__ __forceinline__ unsigned bf2(float lo, float hi) {
    f32x2 f; f.x = lo; f.y = hi;
    bf16x2 b = __builtin_convertvector(f, bf16x2);
    return __builtin_bit_cast(unsigned, b);
}
__device__ __forceinline__ unsigned short bf1(float x) {
    __bf16 b = (__bf16)x;
    return __builtin_bit_cast(unsigned short, b);
}

__device__ __forceinline__ v8f wmma_bf16(const Frag& a, const Frag& b, v8f c) {
    return __builtin_amdgcn_wmma_f32_16x16x32_bf16(
        false, a.v, false, b.v, (short)0, c, false, false);
}

__device__ __forceinline__ float rmax16(float v) {
#pragma unroll
    for (int off = 8; off; off >>= 1) v = fmaxf(v, __shfl_xor(v, off, 32));
    return v;
}
__device__ __forceinline__ float rsum16(float v) {
#pragma unroll
    for (int off = 8; off; off >>= 1) v += __shfl_xor(v, off, 32);
    return v;
}

#if USE_TDM
// TDM: copy one 64x64 bf16 tile (row-major, rows of 128B) global -> LDS,
// padding +16B per row so LDS row stride is 144B (bank-conflict-free-ish).
__device__ __forceinline__ void tdm_load_w(const void* gsrc, unsigned lds_off) {
    unsigned long long ga = (unsigned long long)(size_t)gsrc;
    v4u g0;
    g0.x = 0x1u;                                  // count=1, user mode
    g0.y = lds_off;                               // lds_addr (byte offset)
    g0.z = (unsigned)(ga & 0xFFFFFFFFu);          // global_addr[31:0]
    g0.w = (unsigned)((ga >> 32) & 0x01FFFFFFu)   // global_addr[56:32]
         | (2u << 30);                            // type = 2 ("image")
    v8i g1;
    g1[0] = (1 << 16)        // data_size = 2 bytes
          | (1 << 20)        // pad_enable
          | (4 << 22)        // pad_interval: 32 dwords (= one 128B row)
          | (3 << 25);       // pad_amount: 4 dwords (16B)
    g1[1] = (KC << 16);      // tensor_dim0 = 64 (bits 79:48, low half)
    g1[2] = (KC << 16);      // tensor_dim0 hi = 0 | tensor_dim1 = 64 (low half)
    g1[3] = (KC << 16);      // tensor_dim1 hi = 0 | tile_dim0 = 64
    g1[4] = KC;              // tile_dim1 = 64 | tile_dim2 = 0
    g1[5] = DHEAD;           // tensor_dim0_stride = 64 (low 32)
    g1[6] = 0;               // stride0 hi | stride1 lo
    g1[7] = 0;
    v4i gz4 = (v4i){0, 0, 0, 0};                  // groups 2/3: dims unused
    v8i gz8 = (v8i){0, 0, 0, 0, 0, 0, 0, 0};      // extra descriptor words (unused)
    __builtin_amdgcn_tensor_load_to_lds(g0, g1, gz4, gz4, gz8, 0);
}
#endif

// ---------------------------------------------------------------------------
// Kernel 0: one-time W f32 -> bf16 (row-major bf16 [k][n], n pairs packed per dword).
// ---------------------------------------------------------------------------
__global__ __launch_bounds__(256) void wprep_kernel(
    const float* __restrict__ Wq, const float* __restrict__ Wk,
    const float* __restrict__ Wv,
    unsigned* __restrict__ Wqb, unsigned* __restrict__ Wkb,
    unsigned* __restrict__ Wvb)
{
    const int i = blockIdx.x * 256 + threadIdx.x;  // pair index, 0..32767
    const float2* q = (const float2*)Wq;
    const float2* k = (const float2*)Wk;
    const float2* v = (const float2*)Wv;
    float2 a = q[i], b = k[i], c = v[i];
    Wqb[i] = bf2(a.x, a.y);
    Wkb[i] = bf2(b.x, b.y);
    Wvb[i] = bf2(c.x, c.y);
}

// ---------------------------------------------------------------------------
// Kernel 1: fused QKV projection + RoPE. W staged via TDM into double-buffered
// LDS; one wave = 16 rows x 64 cols; bf16 WMMA with f32 accumulation.
// ---------------------------------------------------------------------------
__global__ __launch_bounds__(256) void qkv_rope_kernel(
    const float* __restrict__ x,
    const unsigned* __restrict__ Wqb, const unsigned* __restrict__ Wkb,
    const unsigned* __restrict__ Wvb,
    unsigned short* __restrict__ Qb, unsigned short* __restrict__ Ktb,
    unsigned short* __restrict__ Vb)
{
    __shared__ unsigned short Wlds[2][3][KC * WROW];   // 2 x 3 x 9216B = 54KB

    const int lane = threadIdx.x & 31;
    const int wave = threadIdx.x >> 5;
    const int ln   = lane & 15;
    const int h    = lane >> 4;
    const int rowbase = (blockIdx.x * 8 + wave) * 16;  // global row in [0, B*T)
    const int b     = rowbase / SEQ;
    const int tbase = rowbase % SEQ;

    const unsigned* Wsrc[3] = {Wqb, Wkb, Wvb};

    v8f aq[4], ak[4], av[4];
#pragma unroll
    for (int i = 0; i < 4; i++) {
        aq[i] = (v8f){0,0,0,0,0,0,0,0};
        ak[i] = (v8f){0,0,0,0,0,0,0,0};
        av[i] = (v8f){0,0,0,0,0,0,0,0};
    }

    const float* xr = x + (size_t)(rowbase + ln) * CEMB;  // A: lane -> M = lane%16

    // ---- stage issue helper: chunk ci -> buffer buf ----
    auto stage = [&](int ci, int buf) {
#if USE_TDM
        if (wave == 0) {
#pragma unroll
            for (int m = 0; m < 3; m++)
                tdm_load_w((const char*)Wsrc[m] + (size_t)ci * KC * DHEAD * 2,
                           (unsigned)(size_t)&Wlds[buf][m][0]);
        }
#else
        // cooperative fallback: 3 x 64 rows x 8 uint4 segments
        for (int idx = threadIdx.x; idx < 3 * KC * 8; idx += 256) {
            int m   = idx >> 9;
            int rem = idx & 511;
            int row = rem >> 3;
            int seg = rem & 7;
            const unsigned* src = Wsrc[m] + (size_t)(ci * KC + row) * 32 + seg * 4;
            *(uint4*)&Wlds[buf][m][row * WROW + seg * 8] = *(const uint4*)src;
        }
#endif
    };

    stage(0, 0);
    stage(1, 1);

    for (int c = 0; c < NCHUNK; c++) {
        const int buf = c & 1;
#if USE_TDM
        if (wave == 0) {
            if (c + 1 < NCHUNK) __builtin_amdgcn_s_wait_tensorcnt(3);
            else                __builtin_amdgcn_s_wait_tensorcnt(0);
        }
#endif
        __syncthreads();   // buffer `buf` (chunk c) is ready for all waves

#pragma unroll
        for (int ks = 0; ks < 2; ks++) {
            const int k0 = c * KC + ks * 32;
            // A-frag (ISA 16-bit A 16x32 layout): two contiguous 32B groups/lane
            Frag A;
            {
                const float4* g0 = (const float4*)(xr + k0 + 8 * h);
                const float4* g1 = (const float4*)(xr + k0 + 16 + 8 * h);
                float4 a0 = g0[0], a1 = g0[1], a2 = g1[0], a3 = g1[1];
                A.u[0] = bf2(a0.x, a0.y); A.u[1] = bf2(a0.z, a0.w);
                A.u[2] = bf2(a1.x, a1.y); A.u[3] = bf2(a1.z, a1.w);
                A.u[4] = bf2(a2.x, a2.y); A.u[5] = bf2(a2.z, a2.w);
                A.u[6] = bf2(a3.x, a3.y); A.u[7] = bf2(a3.z, a3.w);
            }
            const int kcol = ks * 32;   // B-frag: lane = N column, elements = K (table order)
#pragma unroll
            for (int nt = 0; nt < 4; nt++) {
                const int col = nt * 16 + ln;
                Frag Bq, Bk, Bv;
#pragma unroll
                for (int i = 0; i < 8; i++) {
                    const int kr = kcol + ((i < 4) ? (8 * h + 2 * i) : (16 + 8 * h + 2 * (i - 4)));
                    Bq.u[i] = (unsigned)Wlds[buf][0][kr * WROW + col] | ((unsigned)Wlds[buf][0][(kr + 1) * WROW + col] << 16);
                    Bk.u[i] = (unsigned)Wlds[buf][1][kr * WROW + col] | ((unsigned)Wlds[buf][1][(kr + 1) * WROW + col] << 16);
                    Bv.u[i] = (unsigned)Wlds[buf][2][kr * WROW + col] | ((unsigned)Wlds[buf][2][(kr + 1) * WROW + col] << 16);
                }
                aq[nt] = wmma_bf16(A, Bq, aq[nt]);
                ak[nt] = wmma_bf16(A, Bk, ak[nt]);
                av[nt] = wmma_bf16(A, Bv, av[nt]);
            }
        }

        __syncthreads();   // all waves done reading buffer `buf`
        if (c + 2 < NCHUNK) stage(c + 2, buf);
    }

    // RoPE in C-layout registers: element (m=r+8h, d=16*nt+ln); pair (d, d+32)
    // is (tile nt, tile nt+2) at the same lane & VGPR.
#pragma unroll
    for (int r = 0; r < 8; r++) {
        const float pos = (float)(tbase + r + 8 * h);
#pragma unroll
        for (int nt = 0; nt < 2; nt++) {
            const float j  = (float)(16 * nt + ln);
            const float fr = __powf(10000.f, -j * (2.0f / 64.0f));
            const float th = pos * fr;
            const float c = __cosf(th), s = __sinf(th);
            float q1 = aq[nt][r], q2 = aq[nt + 2][r];
            aq[nt][r]     = q1 * c - q2 * s;
            aq[nt + 2][r] = q2 * c + q1 * s;
            float k1 = ak[nt][r], k2 = ak[nt + 2][r];
            ak[nt][r]     = k1 * c - k2 * s;
            ak[nt + 2][r] = k2 * c + k1 * s;
        }
    }

    unsigned short* Qp = Qb  + (size_t)b * SEQ * DHEAD;
    unsigned short* Kp = Ktb + (size_t)b * DHEAD * SEQ;
    unsigned short* Vp = Vb  + (size_t)b * SEQ * DHEAD;
#pragma unroll
    for (int nt = 0; nt < 4; nt++) {
        const int d = 16 * nt + ln;
#pragma unroll
        for (int r = 0; r < 8; r++) {
            const int t = tbase + r + 8 * h;
            Qp[(size_t)t * DHEAD + d] = bf1(aq[nt][r]);
            Vp[(size_t)t * DHEAD + d] = bf1(av[nt][r]);
        }
#pragma unroll
        for (int r = 0; r < 8; r += 2) {
            const int t = tbase + r + 8 * h;                 // even
            *(unsigned*)(Kp + (size_t)d * SEQ + t) = bf2(ak[nt][r], ak[nt][r + 1]);
        }
    }
}

// ---------------------------------------------------------------------------
// Kernel 2: flash attention. One wave = one 16-row query tile, Q frags kept
// resident; 32 keys / iteration: 4 WMMA for S, online softmax, 4 WMMA for PV.
// ---------------------------------------------------------------------------
__global__ __launch_bounds__(256) void attn_kernel(
    const unsigned short* __restrict__ Qb, const unsigned short* __restrict__ Ktb,
    const unsigned short* __restrict__ Vb, float* __restrict__ out)
{
    __shared__ unsigned short lds_p[8][16][34];  // per-wave P staging, padded rows

    const int lane = threadIdx.x & 31;
    const int wave = threadIdx.x >> 5;
    const int ln   = lane & 15;
    const int h    = lane >> 4;

    const int tile  = blockIdx.x + 128 * wave;   // 0..1023 (gridDim.x == 128)
    const int b     = tile >> 7;
    const int qbase = (tile & 127) * 16;

    const unsigned short* Qp = Qb  + (size_t)b * SEQ * DHEAD;
    const unsigned short* Kp = Ktb + (size_t)b * DHEAD * SEQ;
    const unsigned short* Vp = Vb  + (size_t)b * SEQ * DHEAD;

    Frag qf[2];
    const int qrow = qbase + ln;
#pragma unroll
    for (int c = 0; c < 2; c++) {
        const unsigned short* p0 = Qp + (size_t)qrow * DHEAD + 32 * c + 8 * h;
        qf[c].q[0] = *(const uint4*)(p0);
        qf[c].q[1] = *(const uint4*)(p0 + 16);
    }

    v8f acc[4];
#pragma unroll
    for (int t = 0; t < 4; t++) acc[t] = (v8f){0,0,0,0,0,0,0,0};
    float m_r[8], l_r[8];
#pragma unroll
    for (int r = 0; r < 8; r++) { m_r[r] = -INFINITY; l_r[r] = 0.f; }

    const float scale = 0.03125f;  // CEMB^-0.5 = 1/32 (reference scales by C, not D)

    for (int kb = 0; kb < qbase + 16; kb += 32) {
        __builtin_prefetch(Kp + (size_t)lane * SEQ + kb + 32, 0, 1);
        __builtin_prefetch(Vp + (size_t)(kb + 32 + lane) * DHEAD, 0, 1);

        v8f S[2];
#pragma unroll
        for (int t = 0; t < 2; t++) {
            int cb = kb + 16 * t;
            if (cb > SEQ - 16) cb = SEQ - 16;     // masked tail: clamp in-bounds
            const unsigned short* kp = Kp + cb + ln;
            Frag B0, B1;
#pragma unroll
            for (int i = 0; i < 8; i++) {
                const int kd = (i < 4) ? (8 * h + 2 * i) : (16 + 8 * h + 2 * (i - 4));
                B0.u[i] = (unsigned)kp[(size_t)kd * SEQ] | ((unsigned)kp[(size_t)(kd + 1) * SEQ] << 16);
                B1.u[i] = (unsigned)kp[(size_t)(32 + kd) * SEQ] | ((unsigned)kp[(size_t)(33 + kd) * SEQ] << 16);
            }
            v8f z = (v8f){0,0,0,0,0,0,0,0};
            z    = wmma_bf16(qf[0], B0, z);
            S[t] = wmma_bf16(qf[1], B1, z);
        }

#pragma unroll
        for (int t = 0; t < 2; t++)
#pragma unroll
            for (int r = 0; r < 8; r++) {
                const int mg = qbase + r + 8 * h;
                const int ng = kb + 16 * t + ln;
                float sv = S[t][r] * scale;
                S[t][r] = (ng > mg) ? -INFINITY : sv;
            }

        float alpha[8];
#pragma unroll
        for (int r = 0; r < 8; r++) {
            float mx = rmax16(fmaxf(S[0][r], S[1][r]));
            float mn = fmaxf(m_r[r], mx);
            alpha[r] = __expf(m_r[r] - mn);
            m_r[r]   = mn;
            float p0 = __expf(S[0][r] - mn);
            float p1 = __expf(S[1][r] - mn);
            S[0][r] = p0; S[1][r] = p1;
            l_r[r] = l_r[r] * alpha[r] + rsum16(p0 + p1);
        }
#pragma unroll
        for (int t = 0; t < 4; t++)
#pragma unroll
            for (int r = 0; r < 8; r++) acc[t][r] *= alpha[r];

#pragma unroll
        for (int t = 0; t < 2; t++)
#pragma unroll
            for (int r = 0; r < 8; r++)
                lds_p[wave][r + 8 * h][16 * t + ln] = bf1(S[t][r]);
        asm volatile("s_wait_dscnt 0x0" ::: "memory");
        Frag pf;
#pragma unroll
        for (int i = 0; i < 8; i++) {
            int kk = (i < 4) ? (8 * h + 2 * i) : (16 + 8 * h + 2 * (i - 4));
            pf.u[i] = *(const unsigned*)&lds_p[wave][ln][kk];
        }

#pragma unroll
        for (int t = 0; t < 4; t++) {
            Frag Bv;
#pragma unroll
            for (int i = 0; i < 8; i++) {
                int kr0 = kb + ((i < 4) ? (8 * h + 2 * i) : (16 + 8 * h + 2 * (i - 4)));
                int kr1 = kr0 + 1;
                if (kr0 > SEQ - 1) kr0 = SEQ - 1;
                if (kr1 > SEQ - 1) kr1 = SEQ - 1;
                Bv.u[i] = (unsigned)Vp[(size_t)kr0 * DHEAD + 16 * t + ln] | ((unsigned)Vp[(size_t)kr1 * DHEAD + 16 * t + ln] << 16);
            }
            acc[t] = wmma_bf16(pf, Bv, acc[t]);
        }
    }

    float inv[8];
#pragma unroll
    for (int r = 0; r < 8; r++) inv[r] = 1.0f / l_r[r];
    float* op = out + (size_t)b * SEQ * DHEAD;
#pragma unroll
    for (int t = 0; t < 4; t++) {
        const int d = 16 * t + ln;
#pragma unroll
        for (int r = 0; r < 8; r++) {
            const int m = qbase + r + 8 * h;
            op[(size_t)m * DHEAD + d] = acc[t][r] * inv[r];
        }
    }
}

extern "C" void kernel_launch(void* const* d_in, const int* in_sizes, int n_in,
                              void* d_out, int out_size, void* d_ws, size_t ws_size,
                              hipStream_t stream) {
    const float* x  = (const float*)d_in[0];
    const float* Wq = (const float*)d_in[1];
    const float* Wk = (const float*)d_in[2];
    const float* Wv = (const float*)d_in[3];
    float* out = (float*)d_out;

    unsigned short* Qb  = (unsigned short*)d_ws;                      // 2 MB
    unsigned short* Ktb = Qb  + (size_t)BATCH * SEQ * DHEAD;          // 2 MB
    unsigned short* Vb  = Ktb + (size_t)BATCH * DHEAD * SEQ;          // 2 MB
    unsigned* Wqb = (unsigned*)(Vb + (size_t)BATCH * SEQ * DHEAD);    // 128 KB each
    unsigned* Wkb = Wqb + (CEMB * DHEAD / 2);
    unsigned* Wvb = Wkb + (CEMB * DHEAD / 2);

    wprep_kernel<<<dim3(CEMB * DHEAD / 512), dim3(256), 0, stream>>>(
        Wq, Wk, Wv, Wqb, Wkb, Wvb);
    qkv_rope_kernel<<<dim3((BATCH * SEQ) / 128), dim3(256), 0, stream>>>(
        x, Wqb, Wkb, Wvb, Qb, Ktb, Vb);
    attn_kernel<<<dim3(128), dim3(256), 0, stream>>>(Qb, Ktb, Vb, out);
}
